// MultiHeadAttention_80607946211993
// MI455X (gfx1250) — matrix operand order repaired and checked by reading, not compiled, not run on hardware
//
#include <hip/hip_runtime.h>
#include <math.h>

// ---- problem constants (from reference) ----
constexpr int kB  = 2;
constexpr int kS  = 2048;
constexpr int kD  = 2048;
constexpr int kH  = 16;
constexpr int kDH = 128;

typedef __bf16 bf16_t;
typedef __attribute__((ext_vector_type(16))) __bf16        v16bf;
typedef __attribute__((ext_vector_type(8)))  float         v8f;
typedef __attribute__((ext_vector_type(4)))  unsigned int  u32x4;
typedef __attribute__((ext_vector_type(4)))  int           i32x4;

union V16B {
  v16bf v;
  u32x4 q[2];
};

// ---- gfx1250 async global->LDS path (guarded: falls back to sync copy) ----
#if defined(__gfx1250__) && __has_builtin(__builtin_amdgcn_global_load_async_to_lds_b128) && \
    __has_builtin(__builtin_amdgcn_s_wait_asynccnt)
#define ASYNC_LDS 1
#else
#define ASYNC_LDS 0
#endif

// copy 16 bytes (8 bf16) global -> LDS
__device__ inline void cp16(const bf16_t* __restrict__ g, bf16_t* __restrict__ l) {
#if ASYNC_LDS
  __builtin_amdgcn_global_load_async_to_lds_b128(
      (__attribute__((address_space(1))) i32x4*)(void*)g,
      (__attribute__((address_space(3))) i32x4*)(void*)l, 0, 0);
#else
  *(u32x4*)l = *(const u32x4*)g;
#endif
}

__device__ inline void async_wait0() {
#if ASYNC_LDS
  __builtin_amdgcn_s_wait_asynccnt(0);
#endif
}

// load a 16-element bf16 fragment from two 16-byte chunks (LDS)
__device__ inline v16bf load2x16B(const bf16_t* p0, const bf16_t* p1) {
  V16B r;
  r.q[0] = *(const u32x4*)p0;
  r.q[1] = *(const u32x4*)p1;
  return r.v;
}

// ------------------------------------------------------------------
// fp32 -> bf16 conversion (bandwidth trivial vs GEMMs)
// ------------------------------------------------------------------
__global__ void cvt_f32_bf16(const float* __restrict__ src, bf16_t* __restrict__ dst, int n) {
  int i = (blockIdx.x * blockDim.x + threadIdx.x) * 4;
  if (i + 3 < n) {
    float4 f = *(const float4*)(src + i);
    dst[i + 0] = (bf16_t)f.x;
    dst[i + 1] = (bf16_t)f.y;
    dst[i + 2] = (bf16_t)f.z;
    dst[i + 3] = (bf16_t)f.w;
  } else {
    for (; i < n; ++i) dst[i] = (bf16_t)src[i];
  }
}

// ------------------------------------------------------------------
// Y[M,N] = (A[M,K] * W[N,K]^T + bias[N]) * out_scale     (nn.Linear)
// 128x128 block tile, k-step 32, double-buffered LDS (async staging
// on gfx1250), 8 waves: each wave = 32 rows x 64 cols
// (2 A-frags x 4 B-frags -> 8 v_wmma per k-step, 12 ds_load_b128)
// ------------------------------------------------------------------
template <bool OUT_BF16>
__global__ __launch_bounds__(256)
void gemm_bt_bias(const bf16_t* __restrict__ A, const bf16_t* __restrict__ Wt,
                  const float* __restrict__ bias, void* __restrict__ Yv,
                  float out_scale, int M, int N, int K) {
  __shared__ bf16_t As[2][128][40]; // 128x32 tile per buffer, +8 pad
  __shared__ bf16_t Bs[2][128][40];

  const int tid  = threadIdx.x;
  const int lane = tid & 31;
  const int wave = tid >> 5;
  const int m0   = blockIdx.y * 128;
  const int n0   = blockIdx.x * 128;

  const int lr = tid >> 1;        // 0..127 : tile row to load
  const int lh = (tid & 1) * 16;  // 0 or 16 : half of the 32-wide k slice

  const int wm = wave & 3;        // 0..3 : 32-row band
  const int wn = wave >> 2;       // 0..1 : 64-col band

  v8f zero8 = {0, 0, 0, 0, 0, 0, 0, 0};
  v8f acc[2][4];
#pragma unroll
  for (int r = 0; r < 2; ++r)
#pragma unroll
    for (int c = 0; c < 4; ++c) acc[r][c] = zero8;

  const int am  = lane & 15;              // A-frag: M across lanes
  const int akb = (lane < 16) ? 0 : 8;    // K base per half-wave
  const int bn  = lane & 15;              // B-frag: N across lanes
  const int bko = (lane < 16) ? 0 : 8;    // K base per half-wave

  const bf16_t* gA = A  + (size_t)(m0 + lr) * K + lh;
  const bf16_t* gW = Wt + (size_t)(n0 + lr) * K + lh;

  // stage first tile into buffer 0
  cp16(gA,     &As[0][lr][lh]);
  cp16(gA + 8, &As[0][lr][lh + 8]);
  cp16(gW,     &Bs[0][lr][lh]);
  cp16(gW + 8, &Bs[0][lr][lh + 8]);
  async_wait0();
  __syncthreads();

  int buf = 0;
  for (int k0 = 0; k0 < K; k0 += 32) {
    // stage next tile into the other buffer while computing this one
    const int kn = k0 + 32;
    if (kn < K) {
      const int nb = buf ^ 1;
      cp16(gA + kn,     &As[nb][lr][lh]);
      cp16(gA + kn + 8, &As[nb][lr][lh + 8]);
      cp16(gW + kn,     &Bs[nb][lr][lh]);
      cp16(gW + kn + 8, &Bs[nb][lr][lh + 8]);
    }

    v16bf a0 = load2x16B(&As[buf][wm * 32 + am][akb],
                         &As[buf][wm * 32 + am][akb + 16]);
    v16bf a1 = load2x16B(&As[buf][wm * 32 + 16 + am][akb],
                         &As[buf][wm * 32 + 16 + am][akb + 16]);
#pragma unroll
    for (int nt = 0; nt < 4; ++nt) {
      v16bf b = load2x16B(&Bs[buf][wn * 64 + nt * 16 + bn][bko],
                          &Bs[buf][wn * 64 + nt * 16 + bn][bko + 16]);
      acc[0][nt] = __builtin_amdgcn_wmma_f32_16x16x32_bf16(
          false, a0, false, b, (short)0, acc[0][nt], false, false);
      acc[1][nt] = __builtin_amdgcn_wmma_f32_16x16x32_bf16(
          false, a1, false, b, (short)0, acc[1][nt], false, false);
    }

    async_wait0();     // async staging landed (no-op on sync path)
    __syncthreads();   // includes dscnt wait on sync path
    buf ^= 1;
  }

  // epilogue: C layout -> VGPR j holds rows j (lanes 0-15) / j+8 (lanes 16-31)
  const int half8 = (lane >> 4) * 8;
  const int cn    = lane & 15;
#pragma unroll
  for (int r = 0; r < 2; ++r) {
#pragma unroll
    for (int nt = 0; nt < 4; ++nt) {
#pragma unroll
      for (int j = 0; j < 8; ++j) {
        int row = m0 + wm * 32 + r * 16 + j + half8;
        int col = n0 + wn * 64 + nt * 16 + cn;
        float v = (acc[r][nt][j] + bias[col]) * out_scale;
        if (OUT_BF16)
          ((bf16_t*)Yv)[(size_t)row * N + col] = (bf16_t)v;
        else
          ((float*)Yv)[(size_t)row * N + col] = v;
      }
    }
  }
}

// ------------------------------------------------------------------
// Causal flash attention, Q/K/V bf16 in [b, s, h*DH + dh] layout.
// Q is pre-scaled by 1/sqrt(DH) in its projection epilogue.
// Block = 128 query rows (8 waves x 16), key tiles of 32, online softmax.
// Per wave per key tile: 8 WMMA (QK^T) + 8 WMMA (PV).
// ------------------------------------------------------------------
__global__ __launch_bounds__(256)
void flash_attn(const bf16_t* __restrict__ Q, const bf16_t* __restrict__ Kg,
                const bf16_t* __restrict__ Vg, bf16_t* __restrict__ O) {
  __shared__ bf16_t Ks[32][136];     // [key][dh], +8 pad
  __shared__ bf16_t Vt[128][40];     // transposed: [dh][key], +8 pad
  __shared__ bf16_t Ps[8][16][40];   // per-wave P staging (C-layout -> A-layout)

  const int tid  = threadIdx.x;
  const int lane = tid & 31;
  const int wave = tid >> 5;
  const int bh   = blockIdx.y;           // 0 .. B*H-1
  const int b    = bh / kH;
  const int h    = bh % kH;
  const int q0b  = blockIdx.x * 128;
  const int q0   = q0b + wave * 16;

  const size_t headOff = (size_t)b * kS * kD + (size_t)h * kDH;

  const int am    = lane & 15;
  const int akb   = (lane < 16) ? 0 : 8;
  const int bko   = (lane < 16) ? 0 : 8;
  const int half8 = (lane >> 4) * 8;
  const int cn    = lane & 15;

  // Q fragments: 16 rows x 128 dh = 4 A-fragments (K=32 each), kept in VGPRs
  v16bf aq[4];
#pragma unroll
  for (int t = 0; t < 4; ++t) {
    const bf16_t* p = Q + headOff + (size_t)(q0 + am) * kD + t * 32 + akb;
    aq[t] = load2x16B(p, p + 16);
  }

  float mrow[8], lrow[8];
  v8f zero8 = {0, 0, 0, 0, 0, 0, 0, 0};
  v8f oacc[8];
#pragma unroll
  for (int j = 0; j < 8; ++j) { mrow[j] = -1e30f; lrow[j] = 0.0f; }
#pragma unroll
  for (int i = 0; i < 8; ++i) oacc[i] = zero8;

  // cooperative K/V tile load indices
  const int lr   = tid >> 3;         // key row 0..31
  const int lseg = (tid & 7) * 16;   // dh segment

  const int nkt = blockIdx.x * 4 + 4; // causal: key tiles up to block diagonal
  for (int kt = 0; kt < nkt; ++kt) {
    const int k0 = kt * 32;
    __syncthreads();
    {
      const bf16_t* pk = Kg + headOff + (size_t)(k0 + lr) * kD + lseg;
      cp16(pk,     &Ks[lr][lseg]);       // async on gfx1250
      cp16(pk + 8, &Ks[lr][lseg + 8]);
      const bf16_t* pv = Vg + headOff + (size_t)(k0 + lr) * kD + lseg;
      V16B tv;
      tv.q[0] = *(const u32x4*)pv;
      tv.q[1] = *(const u32x4*)(pv + 8);
#pragma unroll
      for (int e = 0; e < 16; ++e) Vt[lseg + e][lr] = tv.v[e];  // transpose into LDS
    }
    async_wait0();
    __syncthreads();

    // scores S = Q K^T  (16 x 32), two 16x16 C fragments
    v8f sc[2];
    sc[0] = zero8; sc[1] = zero8;
#pragma unroll
    for (int t = 0; t < 4; ++t) {
#pragma unroll
      for (int nt = 0; nt < 2; ++nt) {
        v16bf bk = load2x16B(&Ks[nt * 16 + cn][t * 32 + bko],
                             &Ks[nt * 16 + cn][t * 32 + bko + 16]);
        sc[nt] = __builtin_amdgcn_wmma_f32_16x16x32_bf16(
            false, aq[t], false, bk, (short)0, sc[nt], false, false);
      }
    }

    // causal mask: only fragments that can straddle the diagonal (wave-uniform)
#pragma unroll
    for (int nt = 0; nt < 2; ++nt) {
      if (k0 + nt * 16 + 15 > q0) {
#pragma unroll
        for (int j = 0; j < 8; ++j) {
          int qg = q0 + j + half8;
          int kg = k0 + nt * 16 + cn;
          if (kg > qg) sc[nt][j] = -1e30f;
        }
      }
    }

    // online softmax: rows live in half-waves -> width-16 xor reductions
#pragma unroll
    for (int j = 0; j < 8; ++j) {
      float v = fmaxf(sc[0][j], sc[1][j]);
      for (int off = 8; off >= 1; off >>= 1)
        v = fmaxf(v, __shfl_xor(v, off, 16));
      float mnew = fmaxf(mrow[j], v);
      float scl  = __expf(mrow[j] - mnew);
      mrow[j] = mnew;
      float p0 = __expf(sc[0][j] - mnew);
      float p1 = __expf(sc[1][j] - mnew);
      sc[0][j] = p0;
      sc[1][j] = p1;
      float rs = p0 + p1;
      for (int off = 8; off >= 1; off >>= 1)
        rs += __shfl_xor(rs, off, 16);
      lrow[j] = lrow[j] * scl + rs;
#pragma unroll
      for (int d8 = 0; d8 < 8; ++d8) oacc[d8][j] *= scl;
    }

    // stage P (C-layout) into LDS, reload in A-layout (wave-private)
#pragma unroll
    for (int nt = 0; nt < 2; ++nt) {
#pragma unroll
      for (int j = 0; j < 8; ++j)
        Ps[wave][j + half8][nt * 16 + cn] = (bf16_t)sc[nt][j];
    }
    v16bf pa = load2x16B(&Ps[wave][am][akb], &Ps[wave][am][akb + 16]);

    // O += P V : B-fragments are contiguous reads from transposed V tile
#pragma unroll
    for (int d8 = 0; d8 < 8; ++d8) {
      v16bf bv = load2x16B(&Vt[d8 * 16 + cn][bko], &Vt[d8 * 16 + cn][bko + 16]);
      oacc[d8] = __builtin_amdgcn_wmma_f32_16x16x32_bf16(
          false, pa, false, bv, (short)0, oacc[d8], false, false);
    }
  }

  // normalize and store bf16 attention output in [b, s, h*DH + dh] layout
#pragma unroll
  for (int d8 = 0; d8 < 8; ++d8) {
#pragma unroll
    for (int j = 0; j < 8; ++j) {
      int s = q0 + j + half8;
      float v = oacc[d8][j] / lrow[j];
      O[headOff + (size_t)s * kD + d8 * 16 + cn] = (bf16_t)v;
    }
  }
}

// ------------------------------------------------------------------
// host launcher
// ------------------------------------------------------------------
extern "C" void kernel_launch(void* const* d_in, const int* in_sizes, int n_in,
                              void* d_out, int out_size, void* d_ws, size_t ws_size,
                              hipStream_t stream) {
  const float* x  = (const float*)d_in[0];
  // d_in[1] = causal mask (bool) — handled analytically in flash_attn
  const float* wq = (const float*)d_in[2];
  const float* bq = (const float*)d_in[3];
  const float* wk = (const float*)d_in[4];
  const float* bk = (const float*)d_in[5];
  const float* wv = (const float*)d_in[6];
  const float* bv = (const float*)d_in[7];
  const float* wo = (const float*)d_in[8];
  const float* bo = (const float*)d_in[9];

  const size_t MD  = (size_t)kB * kS * kD; // 8,388,608
  const size_t DD2 = (size_t)kD * kD;      // 4,194,304

  bf16_t* ws  = (bf16_t*)d_ws;
  bf16_t* xb  = ws;
  bf16_t* wqb = xb  + MD;
  bf16_t* wkb = wqb + DD2;
  bf16_t* wvb = wkb + DD2;
  bf16_t* wob = wvb + DD2;
  bf16_t* qb  = wob + DD2;
  bf16_t* kbf = qb  + MD;
  bf16_t* vbf = kbf + MD;
  bf16_t* ab  = vbf + MD;

  const int thr = 256;
  {
    int blocks = (int)((MD / 4 + thr - 1) / thr);
    cvt_f32_bf16<<<blocks, thr, 0, stream>>>(x, xb, (int)MD);
    int wblocks = (int)((DD2 / 4 + thr - 1) / thr);
    cvt_f32_bf16<<<wblocks, thr, 0, stream>>>(wq, wqb, (int)DD2);
    cvt_f32_bf16<<<wblocks, thr, 0, stream>>>(wk, wkb, (int)DD2);
    cvt_f32_bf16<<<wblocks, thr, 0, stream>>>(wv, wvb, (int)DD2);
    cvt_f32_bf16<<<wblocks, thr, 0, stream>>>(wo, wob, (int)DD2);
  }

  const float inv_sqrt_dh = 0.08838834764831845f; // 1/sqrt(128), folded into Q proj

  dim3 gg(kD / 128, (kB * kS) / 128); // (16, 32)
  gemm_bt_bias<true><<<gg, 256, 0, stream>>>(xb, wqb, bq, qb,  inv_sqrt_dh, kB * kS, kD, kD);
  gemm_bt_bias<true><<<gg, 256, 0, stream>>>(xb, wkb, bk, kbf, 1.0f,        kB * kS, kD, kD);
  gemm_bt_bias<true><<<gg, 256, 0, stream>>>(xb, wvb, bv, vbf, 1.0f,        kB * kS, kD, kD);

  dim3 ga(kS / 128, kB * kH); // (16, 32)
  flash_attn<<<ga, 256, 0, stream>>>(qb, kbf, vbf, ab);

  gemm_bt_bias<false><<<gg, 256, 0, stream>>>(ab, wob, bo, d_out, 1.0f, kB * kS, kD, kD);
}
